// DotProductAttention_90383291777060
// MI455X (gfx1250) — matrix operand order repaired and checked by reading, not compiled, not run on hardware
//
#include <hip/hip_runtime.h>
#include <hip/hip_bf16.h>
#include <stdint.h>

#define N_BATCH 16
#define CXD 256
#define TXD 2048
#define TYD 2048
#define YB 32
#define SSTR 33          // Slds row stride (floats): 33 -> conflict-free 16-row pattern
#define SCALE 0.0625f    // 1/sqrt(256)

typedef __attribute__((ext_vector_type(16))) __bf16 v16bf;
typedef __attribute__((ext_vector_type(8)))  float  v8f;

__global__ __launch_bounds__(256)
void attn_fused_kernel(const float* __restrict__ K,
                       const float* __restrict__ V,
                       const float* __restrict__ Q,
                       float* __restrict__ Rout,
                       float* __restrict__ Aout)
{
    __shared__ __align__(16) float Slds[TXD * SSTR];  // 264 KB: full score block (f32)
    __shared__ __align__(16) float Qstage[CXD * YB];  // 32 KB: Q block staged f32 (async)
    __shared__ float redmax[8 * YB];
    __shared__ float redsum[8 * YB];
    __shared__ float gmaxs[YB];
    __shared__ float rinvs[YB];

    const int tid  = threadIdx.x;
    const int wave = tid >> 5;
    const int lane = tid & 31;
    const int half = lane >> 4;   // 0: lanes 0-15, 1: lanes 16-31
    const int col  = lane & 15;

    const int n  = blockIdx.y;
    const int y0 = blockIdx.x * YB;

    const float* Kn = K + (size_t)n * CXD * TXD;
    const float* Vn = V + (size_t)n * CXD * TXD;
    const float* Qn = Q + (size_t)n * CXD * TYD;

    // ---- Stage Q block (CXD x YB f32) into LDS with async copies (ASYNCcnt) ----
    {
        const float* qbase = Qn + y0;
        const unsigned lds0 = (unsigned)(uintptr_t)&Qstage[0];
        #pragma unroll
        for (int i = 0; i < 8; ++i) {
            const int cid  = tid + i * 256;       // 2048 chunks of 16B
            const int row  = cid >> 3;
            const int cofs = (cid & 7) * 4;       // float offset in 32-float row
            const float* gsrc = qbase + (size_t)row * TYD + cofs;
            const unsigned dofs = lds0 + (unsigned)((row * YB + cofs) * 4);
            asm volatile("global_load_async_to_lds_b128 %0, %1, off"
                         :: "v"(dofs), "v"(gsrc) : "memory");
        }
        asm volatile("s_wait_asynccnt 0" ::: "memory");
    }
    __syncthreads();

    // ---- Build Q B-fragments for both y-halves (8 c-chunks each) ----
    // B (32x16): lanes 0-15 hold k=0..7,16..23, lanes 16-31 k=8..15,24..31
    v16bf bq0[8], bq1[8];
    #pragma unroll
    for (int cc = 0; cc < 8; ++cc) {
        #pragma unroll
        for (int h = 0; h < 16; ++h) {
            const int c = cc * 32 + half * 8 + ((h < 8) ? h : (h + 8));
            bq0[cc][h] = (__bf16)Qstage[c * YB + col];
            bq1[cc][h] = (__bf16)Qstage[c * YB + col + 16];
        }
    }

    // ---- Phase 1: S = (K^T Q)*scale, rows [wave*256, wave*256+256), 32 cols ----
    float lmax0 = -3.0e38f, lmax1 = -3.0e38f;
    const int tbase = wave * 256;
    for (int it = 0; it < 16; ++it) {
        const int t0 = tbase + it * 16;
        const float* kp = Kn + t0 + col;          // A-matrix row m = t0+col
        __builtin_prefetch(kp + 16, 0, 0);
        v8f acc0 = {0.f,0.f,0.f,0.f,0.f,0.f,0.f,0.f};
        v8f acc1 = {0.f,0.f,0.f,0.f,0.f,0.f,0.f,0.f};
        #pragma unroll
        for (int cc = 0; cc < 8; ++cc) {
            // A (16x32 16-bit): lanes 0-15: k=0..7,16..23 ; lanes 16-31: k=8..15,24..31
            v16bf af;
            const int cbase = cc * 32 + half * 8;
            #pragma unroll
            for (int h = 0; h < 16; ++h) {
                const int k = cbase + ((h < 8) ? 0 : 16) + (h & 7);
                af[h] = (__bf16)kp[(size_t)k * TXD];
            }
            acc0 = __builtin_amdgcn_wmma_f32_16x16x32_bf16(
                false, af, false, bq0[cc], (short)0, acc0, false, false);
            acc1 = __builtin_amdgcn_wmma_f32_16x16x32_bf16(
                false, af, false, bq1[cc], (short)0, acc1, false, false);
        }
        #pragma unroll
        for (int j = 0; j < 8; ++j) {             // C/D: row = j + 8*half
            const float s0 = acc0[j] * SCALE;
            const float s1 = acc1[j] * SCALE;
            lmax0 = fmaxf(lmax0, s0);
            lmax1 = fmaxf(lmax1, s1);
            const int r = t0 + j + 8 * half;
            Slds[r * SSTR + col]      = s0;
            Slds[r * SSTR + col + 16] = s1;
        }
    }

    // ---- Softmax max reduction over Tx (32 y columns) ----
    lmax0 = fmaxf(lmax0, __shfl_xor(lmax0, 16, 32));
    lmax1 = fmaxf(lmax1, __shfl_xor(lmax1, 16, 32));
    if (lane < 16) {
        redmax[wave * YB + lane]      = lmax0;
        redmax[wave * YB + lane + 16] = lmax1;
    }
    __syncthreads();
    if (tid < YB) {
        float m = redmax[tid];
        #pragma unroll
        for (int w2 = 1; w2 < 8; ++w2) m = fmaxf(m, redmax[w2 * YB + tid]);
        gmaxs[tid] = m;
    }
    __syncthreads();

    // ---- Phase 2: exp in place + sum; lane owns y = lane over wave's t-range ----
    {
        const float gm = gmaxs[lane];
        float lsum = 0.f;
        for (int i = 0; i < 256; ++i) {
            const int idx = (tbase + i) * SSTR + lane;
            const float e = __expf(Slds[idx] - gm);
            Slds[idx] = e;
            lsum += e;
        }
        redsum[wave * YB + lane] = lsum;
    }
    __syncthreads();
    if (tid < YB) {
        float s = redsum[tid];
        #pragma unroll
        for (int w2 = 1; w2 < 8; ++w2) s += redsum[w2 * YB + tid];
        rinvs[tid] = 1.0f / s;
    }
    __syncthreads();

    // ---- Phase 3: write A = e * (1/sum), f32, 128B-coalesced rows ----
    {
        const float rv = rinvs[lane];
        float* Abase = Aout + (size_t)n * TXD * TYD + y0;
        for (int i = 0; i < 256; ++i) {
            const int r = tbase + i;
            Abase[(size_t)r * TYD + lane] = Slds[r * SSTR + lane] * rv;
        }
    }

    // ---- Phase 4: R = V * E; wave owns c0=wave*32..+31; V frag reused 2x ----
    const float rv0 = rinvs[col];
    const float rv1 = rinvs[col + 16];
    #pragma unroll
    for (int ct = 0; ct < 2; ++ct) {
        const int cbase = wave * 32 + ct * 16;
        const int c = cbase + col;                // A-matrix row m = channel
        v8f acc0 = {0.f,0.f,0.f,0.f,0.f,0.f,0.f,0.f};
        v8f acc1 = {0.f,0.f,0.f,0.f,0.f,0.f,0.f,0.f};
        for (int t0 = 0; t0 < TXD; t0 += 32) {
            const float* vrow = Vn + (size_t)c * TXD + t0 + half * 8;
            __builtin_prefetch(vrow + 32, 0, 0);
            v16bf af;                             // table-order k: two 8-float runs -> b128 loads
            #pragma unroll
            for (int h = 0; h < 16; ++h) {
                const int k = (h < 8) ? h : (h + 8);
                af[h] = (__bf16)vrow[k];
            }
            v16bf bf0, bf1;                       // B = E(t, y) from LDS (conflict-free)
            #pragma unroll
            for (int h = 0; h < 16; ++h) {
                const int r = t0 + half * 8 + ((h < 8) ? h : (h + 8));
                bf0[h] = (__bf16)Slds[r * SSTR + col];
                bf1[h] = (__bf16)Slds[r * SSTR + col + 16];
            }
            acc0 = __builtin_amdgcn_wmma_f32_16x16x32_bf16(
                false, af, false, bf0, (short)0, acc0, false, false);
            acc1 = __builtin_amdgcn_wmma_f32_16x16x32_bf16(
                false, af, false, bf1, (short)0, acc1, false, false);
        }
        #pragma unroll
        for (int j = 0; j < 8; ++j) {
            const int cr = cbase + j + 8 * half;
            float* rrow = Rout + ((size_t)n * CXD + cr) * TYD + y0;
            rrow[col]      = acc0[j] * rv0;
            rrow[col + 16] = acc1[j] * rv1;
        }
    }
}

extern "C" void kernel_launch(void* const* d_in, const int* in_sizes, int n_in,
                              void* d_out, int out_size, void* d_ws, size_t ws_size,
                              hipStream_t stream) {
    const float* K = (const float*)d_in[0];
    const float* V = (const float*)d_in[1];
    const float* Q = (const float*)d_in[2];
    float* Rout = (float*)d_out;
    float* Aout = Rout + (size_t)N_BATCH * CXD * TYD;  // outputs concatenated: R then A
    dim3 grid(TYD / YB, N_BATCH);
    attn_fused_kernel<<<grid, dim3(256), 0, stream>>>(K, V, Q, Rout, Aout);
}
